// SPINN_38577396253370
// MI455X (gfx1250) — matrix operand order repaired and checked by reading, compiled, not run on hardware
//
#include <hip/hip_runtime.h>
#include <hip/hip_bf16.h>
#include <math.h>

// Problem constants (match reference)
#define B_    128
#define N_    128
#define S_    128
#define TS_   128
#define E_    300
#define EPAD  320      // E padded to multiple of 32 for WMMA K
#define T_    255      // 2N-1 transitions

typedef __attribute__((ext_vector_type(16))) _Float16 v16h;
typedef __attribute__((ext_vector_type(8)))  float    v8f;

__device__ __forceinline__ int   imax(int a, int b) { return a > b ? a : b; }
__device__ __forceinline__ int   imin(int a, int b) { return a < b ? a : b; }
__device__ __forceinline__ float sigm(float x)      { return 1.f / (1.f + expf(-x)); }

// ---------------------------------------------------------------------------
// WMMA fragment loaders per CDNA5 ISA 7.12.2 (wave32, 16-bit operands).
// A: 16x32 (MxK) tile, row-major, leading dim lda.
//   lane l: m = l&15, half = l>>4
//   v0..v3: K = {0,1,2,3,4,5,6,7} + 8*half ; v4..v7: K = {16..23} + 8*half
__device__ __forceinline__ v16h load_a_frag(const _Float16* A, int lda, int lane, int k0) {
  v16h a;
  int m = lane & 15, half = lane >> 4;
#pragma unroll
  for (int r = 0; r < 8; ++r) {
    int k = k0 + ((r < 4) ? (2 * r + 8 * half) : (16 + 2 * (r - 4) + 8 * half));
    a[2 * r]     = A[m * lda + k];
    a[2 * r + 1] = A[m * lda + k + 1];
  }
  return a;
}

// B: 32x16 (KxN). Weights are stored TRANSPOSED: BT[n][k], row-major, ld = ldb.
//   lane l: n = l&15, half = l>>4 ; v0..v3: K = {0..7} + 8*half ; v4..v7: K = {16..23} + 8*half
__device__ __forceinline__ v16h load_b_frag(const _Float16* BT, int ldb, int lane, int k0) {
  v16h b;
  int n = lane & 15, half = lane >> 4;
#pragma unroll
  for (int r = 0; r < 8; ++r) {
    int k = k0 + ((r < 4) ? (2 * r + 8 * half) : (16 + 2 * (r - 4) + 8 * half));
    b[2 * r]     = BT[n * ldb + k];
    b[2 * r + 1] = BT[n * ldb + k + 1];
  }
  return b;
}
// C/D tile (f32): lane l, vgpr r -> D[r + 8*(l>>4)][l&15]

// ---------------------------------------------------------------------------
// Init per-call state (harness does not re-zero between replays).
__global__ void spinn_init(float* h, float* c, int* ptr, int* bptr, float* loss) {
  int idx = blockIdx.x * blockDim.x + threadIdx.x;
  if (idx < B_ * TS_) { h[idx] = 0.f; c[idx] = 0.f; }
  if (idx < B_)       { ptr[idx] = 0; bptr[idx] = 0; }
  if (idx == 0)       *loss = 0.f;
}

// ---------------------------------------------------------------------------
// Pack all GEMM weights into f16, transposed [Nout][Kpad] row-major.
//   Wp_T   [256][320]  <- W_proj   [300][256], K zero-padded 300->320
//   Wlstm_T[512][512]  <- vstack(W_buf, W_s1, W_s2, W_lat) each [128][512]
//   Wred_T [640][384]  <- vstack(W_left, W_right, W_track) each [128][640]
__global__ void spinn_pack_weights(const float* W_proj, const float* W_buf,
                                   const float* W_s1, const float* W_s2,
                                   const float* W_lat, const float* W_left,
                                   const float* W_right, const float* W_track,
                                   _Float16* Wp_T, _Float16* Wlstm_T, _Float16* Wred_T) {
  const int NP = 256 * EPAD;          // 81920
  const int NL = 512 * 512;           // 262144
  const int NR = 640 * 384;           // 245760
  for (int idx = blockIdx.x * blockDim.x + threadIdx.x; idx < NP + NL + NR;
       idx += gridDim.x * blockDim.x) {
    if (idx < NP) {
      int n = idx / EPAD, k = idx % EPAD;
      Wp_T[idx] = (k < E_) ? (_Float16)W_proj[k * 256 + n] : (_Float16)0.f;
    } else if (idx < NP + NL) {
      int j = idx - NP;
      int n = j / 512, k = j % 512;
      const float* src = (k < 128) ? W_buf : (k < 256) ? W_s1 : (k < 384) ? W_s2 : W_lat;
      Wlstm_T[j] = (_Float16)src[(k & 127) * 512 + n];
    } else {
      int j = idx - NP - NL;
      int n = j / 384, k = j % 384;
      const float* src = (k < 128) ? W_left : (k < 256) ? W_right : W_track;
      Wred_T[j] = (_Float16)src[(k & 127) * 640 + n];
    }
  }
}

// ---------------------------------------------------------------------------
// buffer[b,n,:] = embed_table[tokens[b,n]] @ W_proj   -> [16384, 256] f32
// 16-row tile per block; 8 waves, 2 column tiles each; K = 320 (10 chunks).
__global__ void spinn_embed_gemm(const int* tokens, const float* embed_table,
                                 const _Float16* Wp_T, float* buffer) {
  __shared__ _Float16 As[16][EPAD];
  const int tid = threadIdx.x;
  const int r0  = blockIdx.x * 16;

  for (int idx = tid; idx < 16 * EPAD; idx += 256) {
    int rr = idx / EPAD, k = idx - rr * EPAD;
    int tok = tokens[r0 + rr];
    As[rr][k] = (k < E_) ? (_Float16)embed_table[(size_t)tok * E_ + k] : (_Float16)0.f;
  }
  __builtin_prefetch(Wp_T + (size_t)tid * 64, 0, 1);
  __syncthreads();

  const int wave = tid >> 5, lane = tid & 31;
#pragma unroll
  for (int tt = 0; tt < 2; ++tt) {
    int n0 = (wave * 2 + tt) * 16;
    v8f acc = {};
    for (int kc = 0; kc < EPAD / 32; ++kc) {
      v16h a = load_a_frag(&As[0][0], EPAD, lane, kc * 32);
      v16h b = load_b_frag(Wp_T + (size_t)n0 * EPAD, EPAD, lane, kc * 32);
      acc = __builtin_amdgcn_wmma_f32_16x16x32_f16(false, a, false, b, (short)0, acc,
                                                   false, false);
    }
    int n = lane & 15, half = lane >> 4;
#pragma unroll
    for (int r = 0; r < 8; ++r)
      buffer[(size_t)(r0 + r + 8 * half) * 256 + n0 + n] = acc[r];
  }
}

// ---------------------------------------------------------------------------
// Step part 1: gather buf_top/s1/s2 (stage for part 2), tracker-LSTM GEMM
// [16,512]x[512,512] per block, pointwise LSTM -> h,c ; logits + loss accum.
__global__ void spinn_tracker_step(int t, const int* transitions, const float* buffer,
                                   const float* stack_, const float* b_lat,
                                   const float* W_trans, const float* b_trans,
                                   const _Float16* Wlstm_T, float* h, float* c,
                                   float* stage_bt, float* stage_s1, float* stage_s2,
                                   const int* ptr, const int* bptr, float* loss_acc) {
  __shared__ _Float16 As[16][512];
  __shared__ float    gbuf[16][512];
  __shared__ float    hbuf[16][128];
  __shared__ int      sptr[16], sbptr[16];

  const int tid = threadIdx.x;
  const int r0  = blockIdx.x * 16;
  if (tid < 16) { sptr[tid] = ptr[r0 + tid]; sbptr[tid] = bptr[r0 + tid]; }
  __builtin_prefetch(Wlstm_T + (size_t)tid * 1024, 0, 1);
  __syncthreads();

  const int row = tid >> 4, cl = tid & 15;
  const int b   = r0 + row;
  const int p   = sptr[row];
  const bool has1 = (p >= 1), has2 = (p >= 2);
  const float* bt  = buffer + ((size_t)b * N_ + imin(sbptr[row], N_ - 1)) * 256;
  const float* s1p = stack_ + ((size_t)b * N_ + imax(p - 1, 0)) * 256;
  const float* s2p = stack_ + ((size_t)b * N_ + imax(p - 2, 0)) * 256;

  for (int j = cl; j < 256; j += 16) {
    float vb = bt[j];
    float v1 = has1 ? s1p[j] : 0.f;
    float v2 = has2 ? s2p[j] : 0.f;
    stage_bt[b * 256 + j] = vb;
    stage_s1[b * 256 + j] = v1;
    stage_s2[b * 256 + j] = v2;
    if (j < 128) {
      As[row][j]       = (_Float16)vb;
      As[row][128 + j] = (_Float16)v1;
      As[row][256 + j] = (_Float16)v2;
      As[row][384 + j] = (_Float16)h[b * TS_ + j];
    }
  }
  __syncthreads();

  const int wave = tid >> 5, lane = tid & 31;
  for (int tile = wave; tile < 32; tile += 8) {
    int n0 = tile * 16;
    v8f acc = {};
    for (int kc = 0; kc < 16; ++kc) {
      v16h a = load_a_frag(&As[0][0], 512, lane, kc * 32);
      v16h bb = load_b_frag(Wlstm_T + (size_t)n0 * 512, 512, lane, kc * 32);
      acc = __builtin_amdgcn_wmma_f32_16x16x32_f16(false, a, false, bb, (short)0, acc,
                                                   false, false);
    }
    int n = lane & 15, half = lane >> 4;
#pragma unroll
    for (int r = 0; r < 8; ++r) gbuf[r + 8 * half][n0 + n] = acc[r];
  }
  __syncthreads();

  const float bs = (t == 0) ? 0.f : 1.f;   // lateral bias absent on first step
  for (int j = cl; j < 128; j += 16) {
    float av = gbuf[row][j]       + bs * b_lat[j];
    float iv = gbuf[row][128 + j] + bs * b_lat[128 + j];
    float fv = gbuf[row][256 + j] + bs * b_lat[256 + j];
    float ov = gbuf[row][384 + j] + bs * b_lat[384 + j];
    float cv = sigm(fv) * c[b * TS_ + j] + sigm(iv) * tanhf(av);
    float hv = sigm(ov) * tanhf(cv);
    c[b * TS_ + j] = cv;
    h[b * TS_ + j] = hv;
    hbuf[row][j]   = hv;
  }
  __syncthreads();

  if (tid < 16) {
    int bb = r0 + tid;
    float l0 = b_trans[0], l1 = b_trans[1];
    for (int k = 0; k < 128; ++k) {
      float hv = hbuf[tid][k];
      l0 += hv * W_trans[k * 2 + 0];
      l1 += hv * W_trans[k * 2 + 1];
    }
    int   tgt = transitions[bb * T_ + t];
    float m   = fmaxf(l0, l1);
    float lse = m + logf(expf(l0 - m) + expf(l1 - m));
    float lt  = (tgt == 0) ? l0 : l1;
    atomicAdd(loss_acc, -(lt - lse));
  }
}

// ---------------------------------------------------------------------------
// Step part 2: TreeLSTM reduce GEMM [16,384]x[384,640], pointwise, transition
// apply (shift pushes staged buf_top, reduce pops two & pushes composed node).
__global__ void spinn_reduce_step(int t, const int* transitions, const float* b_reduce,
                                  const _Float16* Wred_T, const float* h,
                                  const float* stage_bt, const float* stage_s1,
                                  const float* stage_s2, float* stack_, int* ptr,
                                  int* bptr) {
  __shared__ _Float16 As[16][384];
  __shared__ float    gbuf[16][640];
  __shared__ int      sptr[16];

  const int tid = threadIdx.x;
  const int r0  = blockIdx.x * 16;
  if (tid < 16) sptr[tid] = ptr[r0 + tid];
  __builtin_prefetch(Wred_T + (size_t)tid * 960, 0, 1);
  __syncthreads();

  const int row = tid >> 4, cl = tid & 15;
  const int b   = r0 + row;

  for (int j = cl; j < 128; j += 16) {
    As[row][j]       = (_Float16)stage_s2[b * 256 + j];   // left  = s2 (h part)
    As[row][128 + j] = (_Float16)stage_s1[b * 256 + j];   // right = s1 (h part)
    As[row][256 + j] = (_Float16)h[b * TS_ + j];          // tracker state
  }
  __syncthreads();

  const int wave = tid >> 5, lane = tid & 31;
  for (int tile = wave; tile < 40; tile += 8) {
    int n0 = tile * 16;
    v8f acc = {};
    for (int kc = 0; kc < 12; ++kc) {
      v16h a = load_a_frag(&As[0][0], 384, lane, kc * 32);
      v16h bb = load_b_frag(Wred_T + (size_t)n0 * 384, 384, lane, kc * 32);
      acc = __builtin_amdgcn_wmma_f32_16x16x32_f16(false, a, false, bb, (short)0, acc,
                                                   false, false);
    }
    int n = lane & 15, half = lane >> 4;
#pragma unroll
    for (int r = 0; r < 8; ++r) gbuf[r + 8 * half][n0 + n] = acc[r];
  }
  __syncthreads();

  const int  tr    = transitions[b * T_ + t];
  const bool shift = (tr == 0);
  const int  pold  = sptr[row];
  const int  wpos  = shift ? pold : imax(pold - 2, 0);
  float* dst = stack_ + ((size_t)b * N_ + wpos) * 256;

  for (int j = cl; j < 128; j += 16) {
    float a2 = gbuf[row][j]       + b_reduce[j];
    float i2 = gbuf[row][128 + j] + b_reduce[128 + j];
    float f1 = gbuf[row][256 + j] + b_reduce[256 + j];
    float f2 = gbuf[row][384 + j] + b_reduce[384 + j];
    float o2 = gbuf[row][512 + j] + b_reduce[512 + j];
    float lc = stage_s2[b * 256 + 128 + j];               // left c
    float rc = stage_s1[b * 256 + 128 + j];               // right c
    float cn = tanhf(a2) * sigm(i2) + sigm(f1) * lc + sigm(f2) * rc;
    float hn = sigm(o2) * tanhf(cn);
    dst[j]       = shift ? stage_bt[b * 256 + j]       : hn;
    dst[128 + j] = shift ? stage_bt[b * 256 + 128 + j] : cn;
  }
  if (cl == 0) {
    ptr[b]  = pold + (shift ? 1 : -1);
    bptr[b] = bptr[b] + (shift ? 1 : 0);
  }
}

// ---------------------------------------------------------------------------
// Output: final = stack[b, max(ptr-1,0)]  ([128,256] f32), then loss scalar.
__global__ void spinn_finalize(const float* stack_, const int* ptr, const float* loss,
                               float* out) {
  int idx = blockIdx.x * blockDim.x + threadIdx.x;
  if (idx < B_ * 256) {
    int b = idx >> 8, j = idx & 255;
    int p = imax(ptr[b] - 1, 0);
    out[idx] = stack_[((size_t)b * N_ + p) * 256 + j];
  }
  if (idx == 0) out[B_ * 256] = *loss / (float)(T_ * B_);
}

// ---------------------------------------------------------------------------
extern "C" void kernel_launch(void* const* d_in, const int* in_sizes, int n_in,
                              void* d_out, int out_size, void* d_ws, size_t ws_size,
                              hipStream_t stream) {
  const int*   tokens      = (const int*)d_in[0];
  const int*   transitions = (const int*)d_in[1];
  const float* embed_table = (const float*)d_in[2];
  const float* W_proj      = (const float*)d_in[3];
  const float* W_buf       = (const float*)d_in[4];
  const float* W_s1        = (const float*)d_in[5];
  const float* W_s2        = (const float*)d_in[6];
  const float* W_lat       = (const float*)d_in[7];
  const float* b_lat       = (const float*)d_in[8];
  const float* W_trans     = (const float*)d_in[9];
  const float* b_trans     = (const float*)d_in[10];
  const float* W_left      = (const float*)d_in[11];
  const float* W_right     = (const float*)d_in[12];
  const float* W_track     = (const float*)d_in[13];
  const float* b_reduce    = (const float*)d_in[14];
  float* out = (float*)d_out;

  // Workspace carve-up (256B aligned blocks)
  char* wp = (char*)d_ws;
  auto carve = [&](size_t bytes) -> void* {
    void* r = (void*)wp;
    wp += (bytes + 255) & ~(size_t)255;
    return r;
  };
  float*    buffer   = (float*)carve((size_t)B_ * N_ * 256 * 4);   // 16 MB
  float*    stack_   = (float*)carve((size_t)B_ * N_ * 256 * 4);   // 16 MB
  float*    h        = (float*)carve((size_t)B_ * TS_ * 4);
  float*    c        = (float*)carve((size_t)B_ * TS_ * 4);
  float*    stage_bt = (float*)carve((size_t)B_ * 256 * 4);
  float*    stage_s1 = (float*)carve((size_t)B_ * 256 * 4);
  float*    stage_s2 = (float*)carve((size_t)B_ * 256 * 4);
  int*      ptr      = (int*)carve(B_ * 4);
  int*      bptr     = (int*)carve(B_ * 4);
  float*    loss     = (float*)carve(256);
  _Float16* Wp_T     = (_Float16*)carve((size_t)256 * EPAD * 2);
  _Float16* Wlstm_T  = (_Float16*)carve((size_t)512 * 512 * 2);
  _Float16* Wred_T   = (_Float16*)carve((size_t)640 * 384 * 2);

  // 1) init state (h=c=0, ptr=bptr=0, loss=0)
  spinn_init<<<64, 256, 0, stream>>>(h, c, ptr, bptr, loss);

  // 2) pack weights to f16 transposed
  spinn_pack_weights<<<576, 256, 0, stream>>>(W_proj, W_buf, W_s1, W_s2, W_lat, W_left,
                                              W_right, W_track, Wp_T, Wlstm_T, Wred_T);

  // 3) embedding + projection GEMM: [16384,320]x[320,256] (WMMA f16)
  spinn_embed_gemm<<<(B_ * N_) / 16, 256, 0, stream>>>(tokens, embed_table, Wp_T, buffer);

  // 4) sequential scan: two kernels per transition step (stream order = sync)
  for (int t = 0; t < T_; ++t) {
    spinn_tracker_step<<<B_ / 16, 256, 0, stream>>>(
        t, transitions, buffer, stack_, b_lat, W_trans, b_trans, Wlstm_T, h, c,
        stage_bt, stage_s1, stage_s2, ptr, bptr, loss);
    spinn_reduce_step<<<B_ / 16, 256, 0, stream>>>(
        t, transitions, b_reduce, Wred_T, h, stage_bt, stage_s1, stage_s2, stack_, ptr,
        bptr);
  }

  // 5) emit final stack tops + mean transition cross-entropy
  spinn_finalize<<<128, 256, 0, stream>>>(stack_, ptr, loss, out);
}
